// DAGVAE_68685116998249
// MI455X (gfx1250) — matrix operand order repaired and checked by reading, not compiled, not run on hardware
//
#include <hip/hip_runtime.h>
#include <hip/hip_bf16.h>

// ---------------------------------------------------------------------------
// DAG-VAE forward for MI455X (gfx1250, wave32, WMMA).
// Sizes from reference: B=256, N=128, D_IN=1024, H1=512, H2=256, L=64, SEM_H=32
// ---------------------------------------------------------------------------

#define B_    256
#define N_    128
#define DIN_  1024
#define H1_   512
#define H2_   256
#define L_    64
#define SEMH_ 32
#define MTOT  (B_ * N_)          // 32768 rows for all big GEMMs

typedef __attribute__((ext_vector_type(16))) __bf16 v16bf;
typedef __attribute__((ext_vector_type(8)))  float  v8f;

union ABFrag { uint4 q[2]; v16bf v; };

__device__ __forceinline__ unsigned short f2bf(float f) {
  unsigned int u = __float_as_uint(f);
  u += 0x7fffu + ((u >> 16) & 1u);        // round-to-nearest-even to bf16
  return (unsigned short)(u >> 16);
}

// Load a 16-lane-wide WMMA operand fragment: two 16-byte chunks per lane.
// A-matrix 16x32 bf16: chunk1 at +16 elements. B-matrix 32x16: chunk1 at +16.
__device__ __forceinline__ v16bf load_frag16(const unsigned short* p, int second) {
  ABFrag u;
  u.q[0] = *(const uint4*)(p);
  u.q[1] = *(const uint4*)(p + second);
  return u.v;
}

__device__ __forceinline__ v8f wmma_bf16(v16bf a, v16bf b, v8f c) {
  // D = A x B + C, fp32 accumulate (16x16x32 bf16)
  return __builtin_amdgcn_wmma_f32_16x16x32_bf16(false, a, false, b,
                                                 (short)0, c, false, false);
}

// ---------------------------------------------------------------------------
// Generic GEMM: C[M,N] = act(A[M,K](bf16) * W[N,K]^T(bf16) + bias)
// Block = 128(M) x 64(N), 8 waves (4x2), each wave 32x32 via 2x2 WMMA tiles.
// Fragments loaded straight from global (everything is L2-resident: 192 MB).
// ---------------------------------------------------------------------------
template<int RELU, int WF, int WB>
__global__ __launch_bounds__(256)
void k_gemm(const unsigned short* __restrict__ A,
            const unsigned short* __restrict__ W,
            const float* __restrict__ bias,
            float* __restrict__ Cf,
            unsigned short* __restrict__ Cb,
            int M, int N, int K)
{
  const int lane  = threadIdx.x & 31;
  const int wave  = threadIdx.x >> 5;
  const int mbase = blockIdx.y * 128 + (wave & 3) * 32;
  const int nbase = blockIdx.x * 64  + (wave >> 2) * 32;
  const int r  = lane & 15;
  const int hi = lane >> 4;

  v8f acc00 = {}, acc01 = {}, acc10 = {}, acc11 = {};

  const unsigned short* pa0 = A + (size_t)(mbase + r) * K + hi * 8;
  const unsigned short* pa1 = pa0 + (size_t)16 * K;
  const unsigned short* pb0 = W + (size_t)(nbase + r) * K + hi * 8;
  const unsigned short* pb1 = pb0 + (size_t)16 * K;

  for (int k0 = 0; k0 < K; k0 += 32) {
    __builtin_prefetch(pa0 + k0 + 32, 0, 0);   // global_prefetch for next k-step
    v16bf a0 = load_frag16(pa0 + k0, 16);
    v16bf a1 = load_frag16(pa1 + k0, 16);
    v16bf b0 = load_frag16(pb0 + k0, 16);
    v16bf b1 = load_frag16(pb1 + k0, 16);
    acc00 = wmma_bf16(a0, b0, acc00);
    acc01 = wmma_bf16(a0, b1, acc01);
    acc10 = wmma_bf16(a1, b0, acc10);
    acc11 = wmma_bf16(a1, b1, acc11);
  }

  // C/D layout: VGPR v, lanes 0-15 -> row v, lanes 16-31 -> row v+8; col = lane&15
  const int col = lane & 15;
  const int rs  = (lane >> 4) * 8;
  v8f* ap[4] = { &acc00, &acc01, &acc10, &acc11 };
  for (int i = 0; i < 2; ++i)
    for (int j = 0; j < 2; ++j) {
      v8f a = *ap[i * 2 + j];
      const int gcol = nbase + j * 16 + col;
      const float bv = bias[gcol];
#pragma unroll
      for (int v = 0; v < 8; ++v) {
        float val = a[v] + bv;
        if (RELU) val = fmaxf(val, 0.0f);
        const size_t idx = (size_t)(mbase + i * 16 + rs + v) * N + gcol;
        if (WF) Cf[idx] = val;
        if (WB) Cb[idx] = f2bf(val);
      }
    }
}

// ---------------------------------------------------------------------------
// fp32 -> bf16 convert (grid-stride)
// ---------------------------------------------------------------------------
__global__ void k_cvt(const float* __restrict__ s, unsigned short* __restrict__ d, long n) {
  long i  = (long)blockIdx.x * blockDim.x + threadIdx.x;
  long st = (long)gridDim.x * blockDim.x;
  for (; i < n; i += st) d[i] = f2bf(s[i]);
}

__global__ void k_zero(float* __restrict__ p, long n) {
  long i  = (long)blockIdx.x * blockDim.x + threadIdx.x;
  long st = (long)gridDim.x * blockDim.x;
  for (; i < n; i += st) p[i] = 0.0f;
}

// ---------------------------------------------------------------------------
// Adjacency: adj = ((logits * (1-I)) > 0); also parent counts max(sum,1)
// ---------------------------------------------------------------------------
__global__ __launch_bounds__(128)
void k_adj(const float* __restrict__ logits, float* __restrict__ adj_out,
           float* __restrict__ adjf, float* __restrict__ cnt)
{
  __shared__ float s[128];
  const int i = blockIdx.x, j = threadIdx.x;
  const float v = logits[i * N_ + j];
  const float a = (i != j && v > 0.0f) ? 1.0f : 0.0f;
  adj_out[i * N_ + j] = a;
  adjf[i * N_ + j]    = a;
  s[j] = a;
  __syncthreads();
  for (int st = 64; st > 0; st >>= 1) {
    if (j < st) s[j] += s[j + st];
    __syncthreads();
  }
  if (j == 0) cnt[i] = fmaxf(s[0], 1.0f);
}

// ---------------------------------------------------------------------------
// z = mu + eps*exp(0.5*logvar) (bf16 out); KL reduction via atomics
// ---------------------------------------------------------------------------
__global__ __launch_bounds__(256)
void k_zkl(const float* __restrict__ mu, const float* __restrict__ lv,
           const float* __restrict__ eps, unsigned short* __restrict__ zb,
           float* __restrict__ kl, long n)
{
  __shared__ float red[256];
  long i0 = (long)blockIdx.x * blockDim.x + threadIdx.x;
  long st = (long)gridDim.x * blockDim.x;
  float part = 0.0f;
  for (long i = i0; i < n; i += st) {
    const float m = mu[i], l = lv[i];
    const float el = __expf(l);
    zb[i] = f2bf(m + eps[i] * __expf(0.5f * l));
    part += 1.0f + l - m * m - el;
  }
  red[threadIdx.x] = part;
  __syncthreads();
  for (int s2 = 128; s2 > 0; s2 >>= 1) {
    if (threadIdx.x < s2) red[threadIdx.x] += red[threadIdx.x + s2];
    __syncthreads();
  }
  if (threadIdx.x == 0) atomicAdd(kl, -0.5f * red[0]);
}

// ---------------------------------------------------------------------------
// SEM scan: single workgroup (sequential dependence over topo order).
// feat[256,128] and hh[256,32] staged in LDS (bf16); per-node MLP via WMMA.
// Node weights (12 KB/node) are double-buffered in LDS and streamed with
// GLOBAL_LOAD_ASYNC_TO_LDS_B128 (ASYNCcnt) one step ahead, hiding their
// latency behind the current node's aggregation + WMMA work. Falls back to a
// plain b128 global->LDS copy if the async builtin is unavailable.
// ---------------------------------------------------------------------------
#if __has_builtin(__builtin_amdgcn_global_load_async_to_lds_b128)
#define SEM_ASYNC 1
#else
#define SEM_ASYNC 0
#endif

// Pointer types matching the builtin signature: int __vector(4) in AS1 / AS3.
typedef int v4i32 __attribute__((vector_size(16)));
typedef __attribute__((address_space(1))) v4i32* gas_v4i32_p;
typedef __attribute__((address_space(3))) v4i32* las_v4i32_p;

#define W1_USHORT (SEMH_ * 2 * L_)   // 32*128 = 4096 ushorts (8 KB)
#define W2_USHORT (L_ * SEMH_)       // 64*32  = 2048 ushorts (4 KB)

__global__ __launch_bounds__(256)
void k_sem(const unsigned short* __restrict__ zb, const float* __restrict__ adjf,
           const float* __restrict__ cnt, const int* __restrict__ topo,
           const unsigned short* __restrict__ W1, const float* __restrict__ b1,
           const unsigned short* __restrict__ W2, const float* __restrict__ b2,
           float* __restrict__ tzf, unsigned short* __restrict__ tzb)
{
  __shared__ unsigned short feat[B_ * 2 * L_];                          // 64 KB
  __shared__ unsigned short hh[B_ * SEMH_];                             // 16 KB
  __shared__ __attribute__((aligned(16)))
      unsigned short wbuf[2][W1_USHORT + W2_USHORT];                    // 24 KB

  const int t = threadIdx.x, lane = t & 31, wave = t >> 5;
  const int r = lane & 15, hi = lane >> 4;

  // Stream one node's W1+W2 (12 KB) into LDS buffer `buf` as 16 B chunks.
  auto issue_weights = [&](int nnode, int buf) {
    const unsigned short* g1 = W1 + (size_t)nnode * W1_USHORT;
    const unsigned short* g2 = W2 + (size_t)nnode * W2_USHORT;
    unsigned short* l1 = &wbuf[buf][0];
    unsigned short* l2 = &wbuf[buf][W1_USHORT];
#if SEM_ASYNC
    for (int c = t; c < W1_USHORT / 8; c += 256)
      __builtin_amdgcn_global_load_async_to_lds_b128(
          (gas_v4i32_p)(g1 + c * 8), (las_v4i32_p)(l1 + c * 8), 0, 0);
    for (int c = t; c < W2_USHORT / 8; c += 256)
      __builtin_amdgcn_global_load_async_to_lds_b128(
          (gas_v4i32_p)(g2 + c * 8), (las_v4i32_p)(l2 + c * 8), 0, 0);
#else
    for (int c = t; c < W1_USHORT / 8; c += 256)
      *(uint4*)(l1 + c * 8) = *(const uint4*)(g1 + c * 8);
    for (int c = t; c < W2_USHORT / 8; c += 256)
      *(uint4*)(l2 + c * 8) = *(const uint4*)(g2 + c * 8);
#endif
  };

  // Preload step 0 weights.
  issue_weights(topo[0], 0);
  asm volatile("s_wait_asynccnt 0" ::: "memory");
  __syncthreads();

  int cur = 0;
  for (int step = 0; step < N_; ++step) {
    const int node = topo[step];
    const float inv = 1.0f / cnt[node];
    const float* arow = adjf + node * N_;

    // Prefetch next node's weights into the other buffer (overlaps everything
    // below; completion enforced at the iteration tail).
    if (step + 1 < N_) issue_weights(topo[step + 1], cur ^ 1);

    // feat = [ z[:,node] | mean over parents of tz ]
    for (int idx = t; idx < B_ * L_; idx += 256) {
      const int b = idx >> 6, l = idx & 63;
      float s = 0.0f;
      for (int p = 0; p < N_; ++p)
        if (arow[p] != 0.0f) s += tzf[((size_t)b * N_ + p) * L_ + l];
      feat[b * 128 + l]      = zb[((size_t)b * N_ + node) * L_ + l];
      feat[b * 128 + 64 + l] = f2bf(s * inv);
    }
    __syncthreads();

    // GEMM1: hh = relu(feat[256,128] x W1[node][32,128]^T + b1) -> LDS bf16
    {
      const unsigned short* W1l = &wbuf[cur][0];
      const int mbase = wave * 32;
      v8f a00 = {}, a01 = {}, a10 = {}, a11 = {};
      for (int k0 = 0; k0 < 128; k0 += 32) {
        v16bf af0 = load_frag16(&feat[(mbase + r) * 128 + k0 + hi * 8], 16);
        v16bf af1 = load_frag16(&feat[(mbase + 16 + r) * 128 + k0 + hi * 8], 16);
        v16bf bf0 = load_frag16(W1l + (size_t)(r) * 128 + k0 + hi * 8, 16);
        v16bf bf1 = load_frag16(W1l + (size_t)(16 + r) * 128 + k0 + hi * 8, 16);
        a00 = wmma_bf16(af0, bf0, a00);
        a01 = wmma_bf16(af0, bf1, a01);
        a10 = wmma_bf16(af1, bf0, a10);
        a11 = wmma_bf16(af1, bf1, a11);
      }
      const int col = lane & 15, rs = (lane >> 4) * 8;
      const float* b1n = b1 + node * SEMH_;
      v8f* ap[4] = { &a00, &a01, &a10, &a11 };
      for (int i = 0; i < 2; ++i)
        for (int j = 0; j < 2; ++j) {
          v8f a = *ap[i * 2 + j];
          const float bv = b1n[j * 16 + col];
#pragma unroll
          for (int v = 0; v < 8; ++v) {
            const float val = fmaxf(a[v] + bv, 0.0f);
            hh[(mbase + i * 16 + rs + v) * SEMH_ + j * 16 + col] = f2bf(val);
          }
        }
    }
    __syncthreads();

    // GEMM2: out = hh[256,32] x W2[node][64,32]^T + b2 -> tz[:,node,:]
    {
      const unsigned short* W2l = &wbuf[cur][W1_USHORT];
      const int mbase = wave * 32;
      v16bf af0 = load_frag16(&hh[(mbase + r) * SEMH_ + hi * 8], 16);
      v16bf af1 = load_frag16(&hh[(mbase + 16 + r) * SEMH_ + hi * 8], 16);
      v8f acc[2][4] = {};
#pragma unroll
      for (int j = 0; j < 4; ++j) {
        v16bf bf = load_frag16(W2l + (size_t)(j * 16 + r) * SEMH_ + hi * 8, 16);
        acc[0][j] = wmma_bf16(af0, bf, acc[0][j]);
        acc[1][j] = wmma_bf16(af1, bf, acc[1][j]);
      }
      const int col = lane & 15, rs = (lane >> 4) * 8;
      const float* b2n = b2 + node * L_;
      for (int i = 0; i < 2; ++i)
        for (int j = 0; j < 4; ++j) {
          const float bv = b2n[j * 16 + col];
#pragma unroll
          for (int v = 0; v < 8; ++v) {
            const float val = acc[i][j][v] + bv;
            const size_t row = (size_t)(mbase + i * 16 + rs + v);  // b index
            const size_t o = (row * N_ + node) * L_ + j * 16 + col;
            tzf[o] = val;
            tzb[o] = f2bf(val);
          }
        }
    }
    // Order global tz stores before next iteration's parent reads, and make
    // sure the prefetched weights for step+1 have landed in LDS.
    asm volatile("s_wait_storecnt 0" ::: "memory");
    asm volatile("s_wait_asynccnt 0" ::: "memory");
    __syncthreads();
    cur ^= 1;
  }
}

// ---------------------------------------------------------------------------
extern "C" void kernel_launch(void* const* d_in, const int* in_sizes, int n_in,
                              void* d_out, int out_size, void* d_ws, size_t ws_size,
                              hipStream_t stream)
{
  const float* x       = (const float*)d_in[0];
  const float* eps     = (const float*)d_in[1];
  const float* adj_log = (const float*)d_in[2];
  const float* enc_W1  = (const float*)d_in[3];
  const float* enc_b1  = (const float*)d_in[4];
  const float* enc_W2  = (const float*)d_in[5];
  const float* enc_b2  = (const float*)d_in[6];
  const float* mu_W    = (const float*)d_in[7];
  const float* mu_b    = (const float*)d_in[8];
  const float* lv_W    = (const float*)d_in[9];
  const float* lv_b    = (const float*)d_in[10];
  const float* dec_W1  = (const float*)d_in[11];
  const float* dec_b1  = (const float*)d_in[12];
  const float* dec_W2  = (const float*)d_in[13];
  const float* dec_b2  = (const float*)d_in[14];
  const float* dec_W3  = (const float*)d_in[15];
  const float* dec_b3  = (const float*)d_in[16];
  const float* sem_W1  = (const float*)d_in[17];
  const float* sem_b1  = (const float*)d_in[18];
  const float* sem_W2  = (const float*)d_in[19];
  const float* sem_b2  = (const float*)d_in[20];
  const int*   topo    = (const int*)d_in[21];

  float* out   = (float*)d_out;
  float* recon = out;                               // 256*128*1024
  float* mu_o  = out + (size_t)33554432;            // 256*128*64
  float* lv_o  = mu_o + (size_t)2097152;
  float* adj_o = lv_o + (size_t)2097152;            // 128*128
  float* kl_o  = adj_o + (size_t)16384;             // scalar

  // ---- workspace layout (bf16 activations keep everything L2-resident) ----
  char* ws = (char*)d_ws;
  size_t off = 0;
  auto alloc = [&](size_t bytes) {
    void* p = ws + off;
    off += (bytes + 255) & ~(size_t)255;
    return p;
  };
  unsigned short* xb   = (unsigned short*)alloc((size_t)MTOT * DIN_ * 2);
  unsigned short* h1b  = (unsigned short*)alloc((size_t)MTOT * H1_ * 2);
  unsigned short* h2b  = (unsigned short*)alloc((size_t)MTOT * H2_ * 2);
  unsigned short* zbuf = (unsigned short*)alloc((size_t)MTOT * L_ * 2);
  float*          tzf  = (float*)         alloc((size_t)MTOT * L_ * 4);
  unsigned short* tzb  = (unsigned short*)alloc((size_t)MTOT * L_ * 2);
  unsigned short* d1b  = (unsigned short*)alloc((size_t)MTOT * H2_ * 2);
  unsigned short* d2b  = (unsigned short*)alloc((size_t)MTOT * H1_ * 2);
  unsigned short* ew1b = (unsigned short*)alloc((size_t)H1_ * DIN_ * 2);
  unsigned short* ew2b = (unsigned short*)alloc((size_t)H2_ * H1_ * 2);
  unsigned short* muwb = (unsigned short*)alloc((size_t)L_ * H2_ * 2);
  unsigned short* lvwb = (unsigned short*)alloc((size_t)L_ * H2_ * 2);
  unsigned short* dw1b = (unsigned short*)alloc((size_t)H2_ * L_ * 2);
  unsigned short* dw2b = (unsigned short*)alloc((size_t)H1_ * H2_ * 2);
  unsigned short* dw3b = (unsigned short*)alloc((size_t)DIN_ * H1_ * 2);
  unsigned short* sw1b = (unsigned short*)alloc((size_t)N_ * W1_USHORT * 2);
  unsigned short* sw2b = (unsigned short*)alloc((size_t)N_ * W2_USHORT * 2);
  float*          adjf = (float*)alloc((size_t)N_ * N_ * 4);
  float*          cntf = (float*)alloc((size_t)N_ * 4);

  auto cvt = [&](const float* s, unsigned short* d, long n) {
    long blocks = (n + 255) / 256;
    if (blocks > 8192) blocks = 8192;
    k_cvt<<<(int)blocks, 256, 0, stream>>>(s, d, n);
  };

  // adjacency + parent counts
  k_adj<<<N_, N_, 0, stream>>>(adj_log, adj_o, adjf, cntf);

  // bf16 conversions
  cvt(x,      xb,   (long)MTOT * DIN_);
  cvt(enc_W1, ew1b, (long)H1_ * DIN_);
  cvt(enc_W2, ew2b, (long)H2_ * H1_);
  cvt(mu_W,   muwb, (long)L_ * H2_);
  cvt(lv_W,   lvwb, (long)L_ * H2_);
  cvt(dec_W1, dw1b, (long)H2_ * L_);
  cvt(dec_W2, dw2b, (long)H1_ * H2_);
  cvt(dec_W3, dw3b, (long)DIN_ * H1_);
  cvt(sem_W1, sw1b, (long)N_ * W1_USHORT);
  cvt(sem_W2, sw2b, (long)N_ * W2_USHORT);

  // encoder
  k_gemm<1, 0, 1><<<dim3(H1_ / 64, MTOT / 128), 256, 0, stream>>>(
      xb, ew1b, enc_b1, nullptr, h1b, MTOT, H1_, DIN_);
  k_gemm<1, 0, 1><<<dim3(H2_ / 64, MTOT / 128), 256, 0, stream>>>(
      h1b, ew2b, enc_b2, nullptr, h2b, MTOT, H2_, H1_);

  // heads (f32 straight into d_out)
  k_gemm<0, 1, 0><<<dim3(L_ / 64, MTOT / 128), 256, 0, stream>>>(
      h2b, muwb, mu_b, mu_o, nullptr, MTOT, L_, H2_);
  k_gemm<0, 1, 0><<<dim3(L_ / 64, MTOT / 128), 256, 0, stream>>>(
      h2b, lvwb, lv_b, lv_o, nullptr, MTOT, L_, H2_);

  // reparameterize + KL (zero accumulators first)
  k_zero<<<2048, 256, 0, stream>>>(tzf, (long)MTOT * L_);
  k_zero<<<1, 32, 0, stream>>>(kl_o, 1);
  k_zkl<<<2048, 256, 0, stream>>>(mu_o, lv_o, eps, zbuf, kl_o, (long)MTOT * L_);

  // SEM sequential scan (single workgroup — inherently serial over topo order)
  k_sem<<<1, 256, 0, stream>>>(zbuf, adjf, cntf, topo,
                               sw1b, sem_b1, sw2b, sem_b2, tzf, tzb);

  // decoder
  k_gemm<1, 0, 1><<<dim3(H2_ / 64, MTOT / 128), 256, 0, stream>>>(
      tzb, dw1b, dec_b1, nullptr, d1b, MTOT, H2_, L_);
  k_gemm<1, 0, 1><<<dim3(H1_ / 64, MTOT / 128), 256, 0, stream>>>(
      d1b, dw2b, dec_b2, nullptr, d2b, MTOT, H1_, H2_);
  k_gemm<0, 1, 0><<<dim3(DIN_ / 64, MTOT / 128), 256, 0, stream>>>(
      d2b, dw3b, dec_b3, recon, nullptr, MTOT, DIN_, H1_);
}
